// HL_LH_block_78151224918199
// MI455X (gfx1250) — compile-verified, matrix operand order checked by reading, not run on hardware
//
#include <hip/hip_runtime.h>
#include <stdint.h>

// ---------------------------------------------------------------------------
// Problem dimensions (match reference)
// ---------------------------------------------------------------------------
static const int ATOM = 128, BOND = 64, ANGLE = 32, DIH = 16;
static const int HH = 8;
static const int NG = 50000,  EGc = 200000;
static const int NLG = 200000, ELG = 500000;
static const int NFG = 500000, EFG = 1000000;

#define DEVI __device__ __forceinline__

typedef __bf16 v16bf __attribute__((ext_vector_type(16)));
typedef float  v8f   __attribute__((ext_vector_type(8)));

union BFfrag { v16bf v; unsigned u[8]; };

// monotonic encoding of float into uint so atomicMax(u32) == float max
DEVI unsigned fenc(float x) {
  unsigned u = __float_as_uint(x);
  return ((int)u < 0) ? ~u : (u ^ 0x80000000u);
}
DEVI float fdec(unsigned e) {
  unsigned u = ((int)e < 0) ? (e ^ 0x80000000u) : ~e;
  return __uint_as_float(u);
}

// pack two f32 -> packed bf16x2 (truncate) with one v_perm_b32
DEVI unsigned pkbf(float lo, float hi) {
  return __builtin_amdgcn_perm(__float_as_uint(hi), __float_as_uint(lo), 0x07060302u);
}

// load 16 consecutive f32 (16B aligned, fully in-bounds) as bf16 fragment
DEVI void load_frag(const float* __restrict__ p, BFfrag& f) {
  const float4* q = (const float4*)p;
  float4 v0 = q[0], v1 = q[1], v2 = q[2], v3 = q[3];
  f.u[0] = pkbf(v0.x, v0.y); f.u[1] = pkbf(v0.z, v0.w);
  f.u[2] = pkbf(v1.x, v1.y); f.u[3] = pkbf(v1.z, v1.w);
  f.u[4] = pkbf(v2.x, v2.y); f.u[5] = pkbf(v2.z, v2.w);
  f.u[6] = pkbf(v3.x, v3.y); f.u[7] = pkbf(v3.z, v3.w);
}

// ---------------------------------------------------------------------------
// C[M,N] = A[M,K] @ W[N,K]^T (+ bias[N]).
// Preconditions: M%16==0, N%(16*NT)==0, K%32==0.
// One wave computes a 16 x (NT*16) strip; A fragment reused across NT WMMAs.
// ---------------------------------------------------------------------------
template <int NT>
__global__ __launch_bounds__(256)
void k_gemm_bf16_wmma(const float* __restrict__ A, const float* __restrict__ W,
                      const float* __restrict__ bias, float* __restrict__ C,
                      int M, int N, int K)
{
  const int  lane    = threadIdx.x & 31;
  const long wave    = (((long)blockIdx.x * blockDim.x) + threadIdx.x) >> 5;
  const int  ngroups = N / (16 * NT);
  const long mtiles  = M >> 4;
  if (wave >= mtiles * (long)ngroups) return;
  const int mt = (int)(wave / ngroups);
  const int ng = (int)(wave % ngroups);

  const int rA = (mt << 4) + (lane & 15);
  const int kh = (lane >> 4) << 4;           // K sub-range 0 or 16 within step

  v8f acc[NT];
#pragma unroll
  for (int i = 0; i < NT; ++i)
#pragma unroll
    for (int r = 0; r < 8; ++r) acc[i][r] = 0.0f;

  const float* arow = A + (size_t)rA * K + kh;
  for (int k0 = 0; k0 < K; k0 += 32) {
    BFfrag a;
    load_frag(arow + k0, a);
#pragma unroll
    for (int i = 0; i < NT; ++i) {
      const int cB = ((ng * NT + i) << 4) + (lane & 15);
      BFfrag b;
      load_frag(W + (size_t)cB * K + kh + k0, b);
      acc[i] = __builtin_amdgcn_wmma_f32_16x16x32_bf16(
          false, a.v, false, b.v, (short)0, acc[i], false, false);
    }
  }

  const int rbase = (mt << 4) + ((lane >> 4) << 3);
#pragma unroll
  for (int i = 0; i < NT; ++i) {
    const int cn = ((ng * NT + i) << 4) + (lane & 15);
    const float bv = bias ? bias[cn] : 0.0f;
#pragma unroll
    for (int r = 0; r < 8; ++r)
      C[(size_t)(rbase + r) * N + cn] = acc[i][r] + bv;
  }
}

// ---------------------------------------------------------------------------
// Elementwise / fill / pad / concat helpers
// ---------------------------------------------------------------------------
__global__ void k_fill_f32(float* p, long n, float v) {
  long t = (long)blockIdx.x * blockDim.x + threadIdx.x;
  if (t < n) p[t] = v;
}
__global__ void k_fill_u32(unsigned* p, long n, unsigned v) {
  long t = (long)blockIdx.x * blockDim.x + threadIdx.x;
  if (t < n) p[t] = v;
}
__global__ void k_add(const float* __restrict__ a, const float* __restrict__ b,
                      float* __restrict__ o, long n) {
  long t = (long)blockIdx.x * blockDim.x + threadIdx.x;
  if (t < n) o[t] = a[t] + b[t];
}
// out[r, 0:Kin) = in[r, 0:Kin), out[r, Kin:Kpad) = 0
__global__ void k_pad_rows(const float* __restrict__ in, float* __restrict__ out,
                           long R, int Kin, int Kpad) {
  long t = (long)blockIdx.x * blockDim.x + threadIdx.x;
  if (t >= R * (long)Kpad) return;
  long r = t / Kpad; int c = (int)(t % Kpad);
  out[t] = (c < Kin) ? in[r * (size_t)Kin + c] : 0.0f;
}
// out[r, 0:F1) = A[r], out[r, F1:F1+F2) = B[r]
__global__ void k_concat(const float* __restrict__ A, const float* __restrict__ B,
                         float* __restrict__ out, long R, int F1, int F2) {
  long t = (long)blockIdx.x * blockDim.x + threadIdx.x;
  int F = F1 + F2;
  if (t >= R * (long)F) return;
  long r = t / F; int c = (int)(t % F);
  out[t] = (c < F1) ? A[r * (size_t)F1 + c] : B[r * (size_t)F2 + (c - F1)];
}

// ---------------------------------------------------------------------------
// Edge feature + attention logits: one thread per (edge, head)
// ---------------------------------------------------------------------------
__global__ void k_edge_f(const int* __restrict__ src, const int* __restrict__ dst,
                         const float* __restrict__ f_ni, const float* __restrict__ f_nj,
                         const float* __restrict__ f_fij,
                         const float* __restrict__ bias, const float* __restrict__ attn,
                         float* __restrict__ f_out, float* __restrict__ logit,
                         long E, int H, int Fe)
{
  long t = (long)blockIdx.x * blockDim.x + threadIdx.x;
  if (t >= E * (long)H) return;
  long e = t / H; int h = (int)(t % H);
  int s = src[e], d = dst[e];
  size_t hb = (size_t)h * Fe;
  const float* pni = f_ni + ((size_t)s * H) * Fe + hb;
  const float* pnj = f_nj + ((size_t)d * H) * Fe + hb;
  const float* pf  = f_fij + ((size_t)e * H) * Fe + hb;
  const float* pb  = bias + hb;
  const float* pa  = attn + hb;
  float* po = f_out + ((size_t)e * H) * Fe + hb;
  float acc = 0.0f;
  for (int j = 0; j < Fe; ++j) {
    float v = pni[j] + pnj[j] + pf[j] + pb[j];
    v = v > 0.0f ? v : 0.01f * v;       // leaky_relu(0.01)
    po[j] = v;
    acc += v * pa[j];
  }
  logit[t] = acc;
}

// segment max via monotonic-u32 atomicMax
__global__ void k_scatter_max(const int* __restrict__ dst, const float* __restrict__ logit,
                              unsigned* __restrict__ emax, long E, int H)
{
  long t = (long)blockIdx.x * blockDim.x + threadIdx.x;
  if (t >= E * (long)H) return;
  long e = t / H; int h = (int)(t % H);
  atomicMax(&emax[(size_t)dst[e] * H + h], fenc(logit[t]));
}

// ex = exp(logit - max[dst]) (in-place), atomic segment sum
__global__ void k_exp_sum(const int* __restrict__ dst, float* __restrict__ logit,
                          const unsigned* __restrict__ emax, float* __restrict__ esum,
                          long E, int H)
{
  long t = (long)blockIdx.x * blockDim.x + threadIdx.x;
  if (t >= E * (long)H) return;
  long e = t / H; int h = (int)(t % H);
  size_t di = (size_t)dst[e] * H + h;
  float ex = __expf(logit[t] - fdec(emax[di]));
  logit[t] = ex;
  atomicAdd(&esum[di], ex);
}

// node_out[d, c] += sum_h (ex/esum) * hfeat[s, h*Fn + c]   (head-summed output)
__global__ void k_aggregate(const int* __restrict__ src, const int* __restrict__ dst,
                            const float* __restrict__ hfeat,
                            const float* __restrict__ exw, const float* __restrict__ esum,
                            float* __restrict__ node_out, long E, int H, int Fn)
{
  long t = (long)blockIdx.x * blockDim.x + threadIdx.x;
  if (t >= E * (long)Fn) return;
  long e = t / Fn; int c = (int)(t % Fn);
  int s = src[e], d = dst[e];
  const float* hp = hfeat + (size_t)s * H * Fn + c;
  const float* ep = exw + (size_t)e * H;
  const float* sp = esum + (size_t)d * H;
  float acc = 0.0f;
  for (int h = 0; h < H; ++h)
    acc += (ep[h] / sp[h]) * hp[(size_t)h * Fn];
  atomicAdd(&node_out[(size_t)d * Fn + c], acc);
}

// edge_out[e, c] = sum_h f[e, h*Fe + c]; rows written with stride `ostride`
// (pad region c in [Fe, ostride) is written with 0)
__global__ void k_headsum(const float* __restrict__ f, float* __restrict__ out,
                          long E, int H, int Fe, int ostride)
{
  long t = (long)blockIdx.x * blockDim.x + threadIdx.x;
  if (t >= E * (long)ostride) return;
  long e = t / ostride; int c = (int)(t % ostride);
  float acc = 0.0f;
  if (c < Fe) {
    const float* p = f + (size_t)e * H * Fe + c;
    for (int h = 0; h < H; ++h) acc += p[(size_t)h * Fe];
  }
  out[t] = acc;
}

// ---------------------------------------------------------------------------
// Host-side orchestration
// ---------------------------------------------------------------------------
struct Bump {
  char* base; size_t off;
  float*    f32(size_t n) { size_t o = (off + 255) & ~(size_t)255; off = o + n * 4; return (float*)(base + o); }
  unsigned* u32(size_t n) { size_t o = (off + 255) & ~(size_t)255; off = o + n * 4; return (unsigned*)(base + o); }
};

static inline unsigned gdim(long n, int blk) { return (unsigned)((n + blk - 1) / blk); }

static void gemm(hipStream_t s, const float* A, const float* W, const float* bias,
                 float* C, int M, int N, int K)
{
  if ((N & 63) == 0) {
    long waves = (long)(M >> 4) * (N >> 6);
    k_gemm_bf16_wmma<4><<<gdim(waves * 32, 256), 256, 0, s>>>(A, W, bias, C, M, N, K);
  } else {  // N == 32
    long waves = (long)(M >> 4) * (N >> 5);
    k_gemm_bf16_wmma<2><<<gdim(waves * 32, 256), 256, 0, s>>>(A, W, bias, C, M, N, K);
  }
}

struct EgatP {
  const float *W_node, *b_node, *W_ni, *W_nj, *W_fij, *attn, *bias;
};
static EgatP getp(void* const* d_in, int base) {
  EgatP p;
  p.W_node = (const float*)d_in[base + 0];
  p.b_node = (const float*)d_in[base + 1];
  p.W_ni   = (const float*)d_in[base + 2];
  p.W_nj   = (const float*)d_in[base + 3];
  p.W_fij  = (const float*)d_in[base + 4];
  p.attn   = (const float*)d_in[base + 5];
  p.bias   = (const float*)d_in[base + 6];
  return p;
}

// full EGAT layer; outputs head-summed node_out [N,Fn] and edge_out
// [E, edge_ostride] (first Fe cols valid, rest zero).  efeat rows have
// length FinE (which must be %32==0; caller pre-pads if needed).
// Bump is taken by value so per-layer scratch is recycled automatically.
static void run_egat(hipStream_t s, Bump ws,
                     const int* src, const int* dst, int E, int N,
                     const float* nfeat, const float* efeat,
                     int FinN, int FinE, int Fn, int Fe, const EgatP& p,
                     float* node_out, float* edge_out, int edge_ostride)
{
  const int H = HH;
  const int HFe = H * Fe, HFn = H * Fn;
  float* f_ni  = ws.f32((size_t)N * HFe);
  float* f_nj  = ws.f32((size_t)N * HFe);
  float* hfeat = ws.f32((size_t)N * HFn);
  float* f_fij = ws.f32((size_t)E * HFe);
  float* f     = ws.f32((size_t)E * HFe);
  float* logit = ws.f32((size_t)E * H);
  unsigned* emax = ws.u32((size_t)N * H);
  float* esum  = ws.f32((size_t)N * H);

  // dense projections on the matrix pipe (bf16 WMMA, f32 accumulate)
  gemm(s, nfeat, p.W_ni,   nullptr,  f_ni,  N, HFe, FinN);
  gemm(s, nfeat, p.W_nj,   nullptr,  f_nj,  N, HFe, FinN);
  gemm(s, nfeat, p.W_node, p.b_node, hfeat, N, HFn, FinN);
  gemm(s, efeat, p.W_fij,  nullptr,  f_fij, E, HFe, FinE);

  long EH = (long)E * H;
  k_edge_f<<<gdim(EH, 256), 256, 0, s>>>(src, dst, f_ni, f_nj, f_fij,
                                         p.bias, p.attn, f, logit, E, H, Fe);

  const unsigned NEG_INF_ENC = 0x007FFFFFu;   // fenc(-inf)
  long NHn = (long)N * H;
  k_fill_u32<<<gdim(NHn, 256), 256, 0, s>>>(emax, NHn, NEG_INF_ENC);
  k_fill_f32<<<gdim(NHn, 256), 256, 0, s>>>(esum, NHn, 0.0f);
  k_scatter_max<<<gdim(EH, 256), 256, 0, s>>>(dst, logit, emax, E, H);
  k_exp_sum<<<gdim(EH, 256), 256, 0, s>>>(dst, logit, emax, esum, E, H);

  long NFnL = (long)N * Fn;
  k_fill_f32<<<gdim(NFnL, 256), 256, 0, s>>>(node_out, NFnL, 0.0f);
  long EFnL = (long)E * Fn;
  k_aggregate<<<gdim(EFnL, 256), 256, 0, s>>>(src, dst, hfeat, logit, esum,
                                              node_out, E, H, Fn);
  long EOL = (long)E * edge_ostride;
  k_headsum<<<gdim(EOL, 256), 256, 0, s>>>(f, edge_out, E, H, Fe, edge_ostride);
}

extern "C" void kernel_launch(void* const* d_in, const int* in_sizes, int n_in,
                              void* d_out, int out_size, void* d_ws, size_t ws_size,
                              hipStream_t stream)
{
  (void)in_sizes; (void)n_in; (void)out_size; (void)ws_size;
  const int*   g_src  = (const int*)d_in[0];
  const int*   g_dst  = (const int*)d_in[1];
  const int*   lg_src = (const int*)d_in[2];
  const int*   lg_dst = (const int*)d_in[3];
  const int*   fg_src = (const int*)d_in[4];
  const int*   fg_dst = (const int*)d_in[5];
  const float* gN_in  = (const float*)d_in[6];
  const float* gE_in  = (const float*)d_in[7];
  const float* lgN_in = (const float*)d_in[8];
  const float* lgE_in = (const float*)d_in[9];
  const float* fgN_in = (const float*)d_in[10];
  const float* fgE_in = (const float*)d_in[11];

  EgatP b1_lg = getp(d_in, 12);
  EgatP b1_g  = getp(d_in, 19);
  EgatP b2_lg = getp(d_in, 26);
  EgatP b2_fg = getp(d_in, 33);
  const float* lg_lin_W = (const float*)d_in[40];
  const float* lg_lin_b = (const float*)d_in[41];
  const float* fg_lin_W = (const float*)d_in[42];
  const float* fg_lin_b = (const float*)d_in[43];

  // output slices (reference return order)
  float* out = (float*)d_out;
  float* o_gN  = out;                               // [NG, ATOM]
  float* o_gE  = o_gN  + (size_t)NG  * ATOM;        // [EG, BOND]
  float* o_lgN = o_gE  + (size_t)EGc * BOND;        // [NLG, BOND]
  float* o_lgE = o_lgN + (size_t)NLG * BOND;        // [ELG, ANGLE]
  float* o_fgN = o_lgE + (size_t)ELG * ANGLE;       // [NFG, ANGLE]
  float* o_fgE = o_fgN + (size_t)NFG * ANGLE;       // [EFG, DIH]

  // persistent ping-pong buffers
  Bump ws{(char*)d_ws, 0};
  float* lgN_a   = ws.f32((size_t)NLG * BOND);
  float* lgN_b   = ws.f32((size_t)NLG * BOND);
  float* lgE_a   = ws.f32((size_t)ELG * ANGLE);
  float* lgE_b   = ws.f32((size_t)ELG * ANGLE);
  float* gN_a    = ws.f32((size_t)NG  * ATOM);
  float* gE_a    = ws.f32((size_t)EGc * BOND);
  float* gE_b    = ws.f32((size_t)EGc * BOND);
  float* fgN_a   = ws.f32((size_t)NFG * ANGLE);
  float* fgN_b   = ws.f32((size_t)NFG * ANGLE);
  float* fgE_p0  = ws.f32((size_t)EFG * 32);        // fg edge feats, padded K=32
  float* fgE_p1  = ws.f32((size_t)EFG * 32);
  float* Wfij_fg = ws.f32((size_t)(DIH * HH) * 32); // b2_fg.W_fij padded [128,32]
  float* catb    = ws.f32((size_t)NFG * (ANGLE * 2)); // max of both concat uses

  // ---------------- block 1 ----------------
  // lg EGAT (b1_lg), 1 iteration
  run_egat(stream, ws, lg_src, lg_dst, ELG, NLG, lgN_in, lgE_in,
           BOND, ANGLE, BOND, ANGLE, b1_lg, lgN_a, lgE_a, ANGLE);
  // g_edge = lg_node + g_edge   (NLG == EG)
  long nAdd = (long)EGc * BOND;
  k_add<<<gdim(nAdd, 256), 256, 0, stream>>>(lgN_a, gE_in, gE_a, nAdd);
  // g EGAT (b1_g), 2 iterations, shared weights
  run_egat(stream, ws, g_src, g_dst, EGc, NG, gN_in, gE_a,
           ATOM, BOND, ATOM, BOND, b1_g, gN_a, gE_b, BOND);
  run_egat(stream, ws, g_src, g_dst, EGc, NG, gN_a, gE_b,
           ATOM, BOND, ATOM, BOND, b1_g, o_gN, o_gE, BOND);

  // ---------------- block 2 ----------------
  // lg_node = concat(lg_node, g_edge) @ lg_lin_W^T + b    [NLG,128] -> [NLG,64]
  long nCat1 = (long)NLG * (BOND * 2);
  k_concat<<<gdim(nCat1, 256), 256, 0, stream>>>(lgN_a, o_gE, catb, NLG, BOND, BOND);
  gemm(stream, catb, lg_lin_W, lg_lin_b, lgN_b, NLG, BOND, BOND * 2);
  // lg EGAT (b2_lg), 2 iterations
  run_egat(stream, ws, lg_src, lg_dst, ELG, NLG, lgN_b, lgE_a,
           BOND, ANGLE, BOND, ANGLE, b2_lg, lgN_a, lgE_b, ANGLE);
  run_egat(stream, ws, lg_src, lg_dst, ELG, NLG, lgN_a, lgE_b,
           BOND, ANGLE, BOND, ANGLE, b2_lg, o_lgN, o_lgE, ANGLE);

  // fg_node = concat(fg_node, lg_edge) @ fg_lin_W^T + b   [NFG,64] -> [NFG,32]
  long nCat2 = (long)NFG * (ANGLE * 2);
  k_concat<<<gdim(nCat2, 256), 256, 0, stream>>>(fgN_in, o_lgE, catb, NFG, ANGLE, ANGLE);
  gemm(stream, catb, fg_lin_W, fg_lin_b, fgN_a, NFG, ANGLE, ANGLE * 2);

  // pad fg W_fij [128,16] -> [128,32] and fg edge input [EFG,16] -> [EFG,32]
  long nWp = (long)(DIH * HH) * 32;
  k_pad_rows<<<gdim(nWp, 256), 256, 0, stream>>>(b2_fg.W_fij, Wfij_fg, DIH * HH, DIH, 32);
  long nEp = (long)EFG * 32;
  k_pad_rows<<<gdim(nEp, 256), 256, 0, stream>>>(fgE_in, fgE_p0, EFG, DIH, 32);
  EgatP b2_fg_pad = b2_fg; b2_fg_pad.W_fij = Wfij_fg;

  // fg EGAT (b2_fg), 3 iterations; intermediate edge feats kept padded (K=32)
  run_egat(stream, ws, fg_src, fg_dst, EFG, NFG, fgN_a, fgE_p0,
           ANGLE, 32, ANGLE, DIH, b2_fg_pad, fgN_b, fgE_p1, 32);
  run_egat(stream, ws, fg_src, fg_dst, EFG, NFG, fgN_b, fgE_p1,
           ANGLE, 32, ANGLE, DIH, b2_fg_pad, fgN_a, fgE_p0, 32);
  run_egat(stream, ws, fg_src, fg_dst, EFG, NFG, fgN_a, fgE_p0,
           ANGLE, 32, ANGLE, DIH, b2_fg_pad, o_fgN, o_fgE, DIH);
}
